// xLSTMBranch_309237645813
// MI455X (gfx1250) — matrix operand order repaired and checked by reading, not compiled, not run on hardware
//
#include <hip/hip_runtime.h>
#include <hip/hip_bf16.h>
#include <cstdint>
#include <cstddef>

// ---------------------------------------------------------------------------
// Problem constants (match reference)
// ---------------------------------------------------------------------------
#define BB   256
#define TT   1024
#define DD   37
#define HH   128
#define GG   512          // 4*H
#define OUTN 128

typedef __attribute__((ext_vector_type(16))) __bf16 v16bf;
typedef __attribute__((ext_vector_type(8)))  __bf16 v8bf;
typedef __attribute__((ext_vector_type(8)))  float  v8f;
typedef __attribute__((ext_vector_type(4)))  unsigned int v4u;
typedef __attribute__((ext_vector_type(8)))  int v8i;
typedef __attribute__((ext_vector_type(4)))  int v4i;

union ABPack { v16bf v; v8bf h[2]; };

// Assemble a v16bf WMMA operand from two 16-byte chunks (works for LDS or global).
__device__ __forceinline__ v16bf load_pair16(const __bf16* p0, const __bf16* p1) {
  ABPack u;
  u.h[0] = *(const v8bf*)p0;
  u.h[1] = *(const v8bf*)p1;
  return u.v;
}

__device__ __forceinline__ float sigf(float x) {
  return 1.0f / (1.0f + __expf(-x));
}
__device__ __forceinline__ float tanh_fast(float x) {
  // tanh(x) = 1 - 2/(e^{2x}+1); saturates correctly for |x| large.
  float e = __expf(2.0f * x);
  return 1.0f - 2.0f / (e + 1.0f);
}

// ---------------------------------------------------------------------------
// Packing kernels: fp32 -> bf16, concatenated weights, combined biases
// ---------------------------------------------------------------------------

// x [B,T,37] fp32 -> sanitized bf16 [B,T,64] (zero-padded K)
__global__ __launch_bounds__(256) void pack_x_kernel(const float* __restrict__ x,
                                                     __bf16* __restrict__ dst) {
  const long total = (long)BB * TT * 64;
  for (long idx = (long)blockIdx.x * blockDim.x + threadIdx.x; idx < total;
       idx += (long)gridDim.x * blockDim.x) {
    const int  k  = (int)(idx & 63);
    const long bt = idx >> 6;
    float v = 0.0f;
    if (k < DD) {
      v = x[bt * DD + k];
      v = (v != v) ? 0.0f : fminf(8.0f, fmaxf(-8.0f, v));   // clip + nan_to_num
    }
    dst[idx] = (__bf16)v;
  }
}

// wcat[n][0:kin] = w_ih[n][:] (zero padded past dsrc), wcat[n][kin:kin+128] = w_hh[n][:]
__global__ __launch_bounds__(256) void pack_wcat_kernel(const float* __restrict__ wih,
                                                        const float* __restrict__ whh,
                                                        __bf16* __restrict__ dst,
                                                        int dsrc, int kin) {
  const int ktot = kin + HH;
  const long total = (long)GG * ktot;
  for (long idx = (long)blockIdx.x * blockDim.x + threadIdx.x; idx < total;
       idx += (long)gridDim.x * blockDim.x) {
    const int n = (int)(idx / ktot);
    const int c = (int)(idx % ktot);
    float v;
    if (c < kin) v = (c < dsrc) ? wih[(long)n * dsrc + c] : 0.0f;
    else         v = whh[(long)n * HH + (c - kin)];
    dst[idx] = (__bf16)v;
  }
}

__global__ __launch_bounds__(256) void add_bias_kernel(const float* __restrict__ a,
                                                       const float* __restrict__ b,
                                                       float* __restrict__ dst, int n) {
  for (int i = blockIdx.x * blockDim.x + threadIdx.x; i < n; i += gridDim.x * blockDim.x)
    dst[i] = a[i] + b[i];
}

__global__ __launch_bounds__(256) void pack_bf16_kernel(const float* __restrict__ src,
                                                        __bf16* __restrict__ dst, int n) {
  for (int i = blockIdx.x * blockDim.x + threadIdx.x; i < n; i += gridDim.x * blockDim.x)
    dst[i] = (__bf16)src[i];
}

// ---------------------------------------------------------------------------
// Persistent LSTM scan kernel.
//   grid = (B/16, 2 directions), block = 256 (8 waves).
//   Wave w owns hidden units [16w, 16w+16); gates for that slice are gate GEMM
//   columns {16w, 128+16w, 256+16w, 384+16w} (+0..15).  Per timestep:
//   gates = bias + [x_t | h_{t-1}] @ wcat^T  via v_wmma_f32_16x16x32_bf16,
//   four independent accumulator chains (i,f,g,o) for XDL pipelining.
// ---------------------------------------------------------------------------
template <int KIN, bool WRITE_SEQ>
__global__ __launch_bounds__(256)
void lstm_scan_kernel(const __bf16* __restrict__ xin,       // [B][T][KIN] bf16
                      const __bf16* __restrict__ wcat_f,    // [512][KIN+128] bf16
                      const __bf16* __restrict__ wcat_r,
                      const float*  __restrict__ bias_f,    // [512] (b_ih+b_hh)
                      const float*  __restrict__ bias_r,
                      __bf16* __restrict__ seqout,          // [B][T][256] (if WRITE_SEQ)
                      __bf16* __restrict__ hfinal,          // [B][512]
                      int hfcol_f, int hfcol_r) {
  constexpr int KTOT = KIN + HH;
  constexpr int NK   = KTOT / 32;   // K-steps of 32 (bf16)
  constexpr int NKX  = KIN / 32;    // K-steps sourcing from x

  const int dir = blockIdx.y;
  const __bf16* __restrict__ wcat = dir ? wcat_r : wcat_f;
  const float*  __restrict__ bias = dir ? bias_r : bias_f;
  const int hfcol = dir ? hfcol_r : hfcol_f;

  const int b0   = blockIdx.x * 16;
  const int tid  = threadIdx.x;
  const int wave = tid >> 5;
  const int lane = tid & 31;
  const int rowm = lane & 15;       // A row / C column index
  const int half = lane >> 4;       // K-half select for A/B layouts
  const int j0   = wave * 16;       // hidden slice base
  const int ncol = rowm;            // N within 16x16 tile (C/D: N = lane%16)

  // h double buffer: [2][16 rows][128 + 8 pad] bf16; row stride 272B (16B mult,
  // 4-bank skew => conflict-free 16B loads).
  __shared__ __align__(16) __bf16 hbuf[2][16][136];
  for (int i = tid; i < 2 * 16 * 136; i += 256) (&hbuf[0][0][0])[i] = (__bf16)0.0f;
  __syncthreads();

  float bg[4];
  #pragma unroll
  for (int g = 0; g < 4; ++g) bg[g] = bias[g * HH + j0 + ncol];

  float cst[8];
  #pragma unroll
  for (int r = 0; r < 8; ++r) cst[r] = 0.0f;

  const long xrow = (long)(b0 + rowm) * TT;

  #pragma unroll 1
  for (int it = 0; it < TT; ++it) {
    const int t   = dir ? (TT - 1 - it) : it;
    const int cur = it & 1;
    const int nxt = cur ^ 1;

    if (it + 1 < TT) {  // prefetch next timestep's activations into cache
      const int tn = dir ? (TT - 2 - it) : (it + 1);
      __builtin_prefetch(xin + (xrow + tn) * KIN, 0, 3);
    }

    v8f acc[4];
    #pragma unroll
    for (int g = 0; g < 4; ++g) {
      #pragma unroll
      for (int r = 0; r < 8; ++r) acc[g][r] = bg[g];
    }

    #pragma unroll
    for (int kk = 0; kk < NK; ++kk) {
      // A operand (16x32 bf16): lane<16 -> K {k0..k0+7, k0+16..k0+23}, k0=kk*32
      //                         lane>=16 -> shifted by 8.
      v16bf a;
      if (kk < NKX) {
        const __bf16* p = xin + (xrow + t) * KIN + kk * 32 + half * 8;
        a = load_pair16(p, p + 16);
      } else {
        const __bf16* p = &hbuf[cur][rowm][(kk - NKX) * 32 + half * 8];
        a = load_pair16(p, p + 16);
      }
      #pragma unroll
      for (int g = 0; g < 4; ++g) {
        // B operand (32x16 bf16): lane holds K {k0+8*half..+7, k0+16+8*half..+7} of column N.
        const __bf16* wp =
            wcat + (long)(g * HH + j0 + ncol) * KTOT + kk * 32 + half * 8;
        v16bf bm = load_pair16(wp, wp + 16);
        acc[g] = __builtin_amdgcn_wmma_f32_16x16x32_bf16(
            false, a, false, bm, (short)0, acc[g], false, false);
      }
    }

    // Gate nonlinearity + state update.  C/D element (vgpr r, lane): M = r+8*half.
    #pragma unroll
    for (int r = 0; r < 8; ++r) {
      const float iv = sigf(acc[0][r]);
      const float fv = sigf(acc[1][r]);
      const float gv = tanh_fast(acc[2][r]);
      const float ov = sigf(acc[3][r]);
      const float c  = fv * cst[r] + iv * gv;
      cst[r] = c;
      float hv = ov * tanh_fast(c);
      if (hv != hv) hv = 0.0f;
      const int m = r + half * 8;
      hbuf[nxt][m][j0 + ncol] = (__bf16)hv;
      if (WRITE_SEQ)
        seqout[((long)(b0 + m) * TT + t) * 256 + dir * HH + j0 + ncol] = (__bf16)hv;
      if (it == TT - 1)
        hfinal[(long)(b0 + m) * 512 + hfcol + j0 + ncol] = (__bf16)hv;
    }
    __syncthreads();
  }
}

// ---------------------------------------------------------------------------
// Projection + LayerNorm + exact GeLU.
//   grid = B/16, block = 256.  z[16,128] = h[16,512] @ w_proj^T + b via WMMA,
//   4 independent split-K accumulator chains to keep the XDL pipe busy.
//   h-tile staged into LDS via Tensor Data Mover (TDM) when available.
// ---------------------------------------------------------------------------
__global__ __launch_bounds__(256)
void proj_ln_gelu_kernel(const __bf16* __restrict__ hf,     // [B][512] bf16
                         const __bf16* __restrict__ wproj,  // [128][512] bf16
                         const float* __restrict__ bproj,
                         const float* __restrict__ gamma,
                         const float* __restrict__ beta,
                         float* __restrict__ out) {          // [B][128] fp32
  const int b0   = blockIdx.x * 16;
  const int tid  = threadIdx.x;
  const int wave = tid >> 5;
  const int lane = tid & 31;
  const int rowm = lane & 15;
  const int half = lane >> 4;
  const int n0   = wave * 16;
  const int ncol = rowm;

  __shared__ __align__(16) __bf16 htile[16][512];
  __shared__ float zt[16][132];
  __shared__ float mu_s[16], rs_s[16];

#if __has_builtin(__builtin_amdgcn_tensor_load_to_lds)
  if (wave == 0) {
    // TDM descriptor: 2D tile 512 x 16 of 2-byte elements, row stride 512.
    const unsigned ldsa = (unsigned)(uintptr_t)(&htile[0][0]);
    const unsigned long long ga =
        (unsigned long long)(uintptr_t)(hf + (long)b0 * 512);
    v4u g0;
    g0[0] = 1u;                                      // count=1, user descriptor
    g0[1] = ldsa;                                    // lds_addr
    g0[2] = (unsigned)(ga & 0xffffffffull);          // global_addr[31:0]
    g0[3] = (unsigned)((ga >> 32) & 0x1ffffffull) | (2u << 30);  // addr hi | type=2
    v8i g1;
    g1[0] = (int)(1u << 16);        // workgroup_mask=0, data_size=1 (2B)
    g1[1] = (int)(512u << 16);      // tensor_dim0 = 512 (low16 at bits[63:48])
    g1[2] = (int)(256u << 16);      // tensor_dim0 hi=0; tensor_dim1 = 256 rows
    g1[3] = (int)(512u << 16);      // tensor_dim1 hi=0; tile_dim0 = 512
    g1[4] = 16;                     // tile_dim1 = 16, tile_dim2 = 0
    g1[5] = 512;                    // tensor_dim0_stride low32
    g1[6] = 0;
    g1[7] = 0;
    v4i z4 = {0, 0, 0, 0};
#if defined(__clang_major__) && (__clang_major__ >= 23)
    v8i z8 = {0, 0, 0, 0, 0, 0, 0, 0};
    __builtin_amdgcn_tensor_load_to_lds(g0, g1, z4, z4, z8, 0);
#else
    __builtin_amdgcn_tensor_load_to_lds(g0, g1, z4, z4, 0);
#endif
    __builtin_amdgcn_s_wait_tensorcnt(0);
  }
  __syncthreads();
#else
  for (int i = tid; i < 16 * 512; i += 256)
    (&htile[0][0])[i] = hf[(long)(b0 + (i >> 9)) * 512 + (i & 511)];
  __syncthreads();
#endif

  // Split-K: 4 independent WMMA accumulator chains (chain c takes kk = c,c+4,...)
  v8f acc[4];
  #pragma unroll
  for (int cchain = 0; cchain < 4; ++cchain) {
    #pragma unroll
    for (int r = 0; r < 8; ++r) acc[cchain][r] = 0.0f;
  }

  #pragma unroll
  for (int kk = 0; kk < 16; ++kk) {
    const __bf16* pa = &htile[rowm][kk * 32 + half * 8];
    v16bf a = load_pair16(pa, pa + 16);
    const __bf16* wp = wproj + (long)(n0 + ncol) * 512 + kk * 32 + half * 8;
    v16bf bm = load_pair16(wp, wp + 16);
    acc[kk & 3] = __builtin_amdgcn_wmma_f32_16x16x32_bf16(
        false, a, false, bm, (short)0, acc[kk & 3], false, false);
  }

  const float bv = bproj[n0 + ncol];
  #pragma unroll
  for (int r = 0; r < 8; ++r)
    zt[r + half * 8][n0 + ncol] =
        bv + (acc[0][r] + acc[1][r]) + (acc[2][r] + acc[3][r]);
  __syncthreads();

  if (tid < 16) {
    float s = 0.0f, ss = 0.0f;
    for (int c = 0; c < OUTN; ++c) {
      const float z = zt[tid][c];
      s += z;
      ss += z * z;
    }
    const float mu = s * (1.0f / OUTN);
    float var = ss * (1.0f / OUTN) - mu * mu;
    var = var < 0.0f ? 0.0f : var;
    mu_s[tid] = mu;
    rs_s[tid] = rsqrtf(var + 1e-5f);
  }
  __syncthreads();

  for (int e = tid; e < 16 * OUTN; e += 256) {
    const int row = e >> 7;
    const int col = e & 127;
    float z = (zt[row][col] - mu_s[row]) * rs_s[row] * gamma[col] + beta[col];
    out[(long)(b0 + row) * OUTN + col] =
        0.5f * z * (1.0f + erff(z * 0.70710678118654752f));
  }
}

// ---------------------------------------------------------------------------
// Host launcher
// ---------------------------------------------------------------------------
extern "C" void kernel_launch(void* const* d_in, const int* in_sizes, int n_in,
                              void* d_out, int out_size, void* d_ws, size_t ws_size,
                              hipStream_t stream) {
  (void)in_sizes; (void)n_in; (void)out_size; (void)ws_size;

  const float* x      = (const float*)d_in[0];
  const float* w_ih0f = (const float*)d_in[1];
  const float* w_hh0f = (const float*)d_in[2];
  const float* b_ih0f = (const float*)d_in[3];
  const float* b_hh0f = (const float*)d_in[4];
  const float* w_ih0r = (const float*)d_in[5];
  const float* w_hh0r = (const float*)d_in[6];
  const float* b_ih0r = (const float*)d_in[7];
  const float* b_hh0r = (const float*)d_in[8];
  const float* w_ih1f = (const float*)d_in[9];
  const float* w_hh1f = (const float*)d_in[10];
  const float* b_ih1f = (const float*)d_in[11];
  const float* b_hh1f = (const float*)d_in[12];
  const float* w_ih1r = (const float*)d_in[13];
  const float* w_hh1r = (const float*)d_in[14];
  const float* b_ih1r = (const float*)d_in[15];
  const float* b_hh1r = (const float*)d_in[16];
  const float* w_proj = (const float*)d_in[17];
  const float* b_proj = (const float*)d_in[18];
  const float* gamma  = (const float*)d_in[19];
  const float* beta   = (const float*)d_in[20];
  float* out = (float*)d_out;

  // Workspace layout (256B aligned slabs)
  uint8_t* ws = (uint8_t*)d_ws;
  size_t off = 0;
  auto take = [&](size_t bytes) {
    size_t r = off;
    off = (off + bytes + 255) & ~(size_t)255;
    return r;
  };
  const size_t o_xbf  = take((size_t)BB * TT * 64 * 2);        // bf16 padded x
  const size_t o_x1   = take((size_t)BB * TT * 256 * 2);       // layer-1 input
  const size_t o_w0f  = take((size_t)GG * 192 * 2);
  const size_t o_w0r  = take((size_t)GG * 192 * 2);
  const size_t o_w1f  = take((size_t)GG * 384 * 2);
  const size_t o_w1r  = take((size_t)GG * 384 * 2);
  const size_t o_b0f  = take((size_t)GG * 4);
  const size_t o_b0r  = take((size_t)GG * 4);
  const size_t o_b1f  = take((size_t)GG * 4);
  const size_t o_b1r  = take((size_t)GG * 4);
  const size_t o_wprj = take((size_t)OUTN * 512 * 2);
  const size_t o_hf   = take((size_t)BB * 512 * 2);

  __bf16* xbf   = (__bf16*)(ws + o_xbf);
  __bf16* x1    = (__bf16*)(ws + o_x1);
  __bf16* w0f   = (__bf16*)(ws + o_w0f);
  __bf16* w0r   = (__bf16*)(ws + o_w0r);
  __bf16* w1f   = (__bf16*)(ws + o_w1f);
  __bf16* w1r   = (__bf16*)(ws + o_w1r);
  float*  bb0f  = (float*)(ws + o_b0f);
  float*  bb0r  = (float*)(ws + o_b0r);
  float*  bb1f  = (float*)(ws + o_b1f);
  float*  bb1r  = (float*)(ws + o_b1r);
  __bf16* wprjb = (__bf16*)(ws + o_wprj);
  __bf16* hf    = (__bf16*)(ws + o_hf);

  // ---- pack / convert ----
  pack_x_kernel<<<8192, 256, 0, stream>>>(x, xbf);
  pack_wcat_kernel<<<512, 256, 0, stream>>>(w_ih0f, w_hh0f, w0f, DD, 64);
  pack_wcat_kernel<<<512, 256, 0, stream>>>(w_ih0r, w_hh0r, w0r, DD, 64);
  pack_wcat_kernel<<<1024, 256, 0, stream>>>(w_ih1f, w_hh1f, w1f, 256, 256);
  pack_wcat_kernel<<<1024, 256, 0, stream>>>(w_ih1r, w_hh1r, w1r, 256, 256);
  add_bias_kernel<<<2, 256, 0, stream>>>(b_ih0f, b_hh0f, bb0f, GG);
  add_bias_kernel<<<2, 256, 0, stream>>>(b_ih0r, b_hh0r, bb0r, GG);
  add_bias_kernel<<<2, 256, 0, stream>>>(b_ih1f, b_hh1f, bb1f, GG);
  add_bias_kernel<<<2, 256, 0, stream>>>(b_ih1r, b_hh1r, bb1r, GG);
  pack_bf16_kernel<<<256, 256, 0, stream>>>(w_proj, wprjb, OUTN * 512);

  // ---- layer 0 (both directions in one launch); final h -> cols 256/384 ----
  lstm_scan_kernel<64, true><<<dim3(BB / 16, 2), 256, 0, stream>>>(
      xbf, w0f, w0r, bb0f, bb0r, x1, hf, /*hfcol_f=*/256, /*hfcol_r=*/384);

  // ---- layer 1; final h -> cols 0/128 ----
  lstm_scan_kernel<256, false><<<dim3(BB / 16, 2), 256, 0, stream>>>(
      x1, w1f, w1r, bb1f, bb1r, (__bf16*)nullptr, hf, /*hfcol_f=*/0,
      /*hfcol_r=*/128);

  // ---- projection + LN + GeLU ----
  proj_ln_gelu_kernel<<<BB / 16, 256, 0, stream>>>(hf, wprjb, b_proj, gamma,
                                                   beta, out);
}
